// ACTORVAE_XT_49048526520737
// MI455X (gfx1250) — matrix operand order repaired and checked by reading, not compiled, not run on hardware
//
#include <hip/hip_runtime.h>
#include <math.h>

typedef __bf16 bf16_t;
typedef __attribute__((ext_vector_type(16))) __bf16 v16bf;
typedef __attribute__((ext_vector_type(8)))  __bf16 v8bf;
typedef __attribute__((ext_vector_type(8)))  float  v8f;

#define NEG_INF (-3.402823466e38f)

// ---------------------------------------------------------------------------
// Generic batched WMMA GEMM, async double-buffered staging.
// Block tile 64(M) x 128(N) x 64(K); 8 waves, each wave computes 32x32 via
// 8x v_wmma_f32_16x16x32_bf16 per staged K tile (2 k-steps of 32).
// Fast paths stage via GLOBAL_LOAD_ASYNC_TO_LDS_B128 (ASYNCcnt, no VGPR
// round-trip); next tile is staged while current tile is computed, waiting
// only for the previous tile via s_wait_asynccnt (async loads complete in
// order). B is staged transposed in LDS so both A and B fragments are
// aligned 16B LDS b128 loads.
// ---------------------------------------------------------------------------
__global__ __launch_bounds__(256)
void wmma_gemm_kernel(const bf16_t* __restrict__ A, const bf16_t* __restrict__ Bm,
                      float* __restrict__ C, bf16_t* __restrict__ Cbf,
                      const float* __restrict__ bias,
                      const float* __restrict__ bmat, int bmat_ld, long long bmat_h,
                      const float* __restrict__ resid,
                      const float* __restrict__ zscale,
                      int M, int N, int K, int lda, int ldb, int ldc,
                      long long sAb, long long sAh, long long sBb, long long sBh,
                      long long sCb, long long sCh, int zmod,
                      float scale, int btrans, int causal)
{
    __shared__ bf16_t As[2][64][72];    // pitch 144B (16B-aligned fragment slots)
    __shared__ bf16_t BsT[2][128][72];  // B stored transposed: BsT[n][k]

    int z  = blockIdx.z;
    int bb = z / zmod, hh = z % zmod;
    A  += (long long)bb * sAb + (long long)hh * sAh;
    Bm += (long long)bb * sBb + (long long)hh * sBh;
    long long coff = (long long)bb * sCb + (long long)hh * sCh;

    int m0 = blockIdx.y * 64;
    int n0 = blockIdx.x * 128;
    int tid  = threadIdx.x;
    int wave = tid >> 5, lane = tid & 31;
    int wm2 = (wave & 1) * 32;        // wave M offset within block tile
    int wn2 = (wave >> 1) * 32;       // wave N offset within block tile
    int half = lane >> 4, l16 = lane & 15;

    const bool fa = ((lda & 7) == 0) && (m0 + 64 <= M);
    const bool fb = ((ldb & 7) == 0) && (n0 + 128 <= N);

    // Stage one 64-deep K tile into buffer `buf`; returns per-thread count of
    // async b128 ops issued (block-uniform).
    auto stageTile = [&](int buf, int k0) -> int {
        int issued = 0;
        const bool inK = (k0 + 64 <= K);
        // ---- A 64x64 ----
        if (fa && inK) {
            #pragma unroll
            for (int i = 0; i < 2; ++i) {
                int idx = tid + i * 256;
                int r = idx >> 3, cg = (idx & 7) * 8;
                unsigned laddr = (unsigned)(unsigned long long)&As[buf][r][cg];
                const bf16_t* ga = &A[(long long)(m0 + r) * lda + k0 + cg];
                asm volatile("global_load_async_to_lds_b128 %0, %1, off"
                             :: "v"(laddr), "v"(ga) : "memory");
            }
            issued += 2;
        } else {
            for (int idx = tid; idx < 64 * 64; idx += 256) {
                int r = idx >> 6, cc = idx & 63;
                int gm = m0 + r, gk = k0 + cc;
                bf16_t v = (bf16_t)0.0f;
                if (gm < M && gk < K) v = A[(long long)gm * lda + gk];
                As[buf][r][cc] = v;
            }
        }
        // ---- B transposed: BsT[n][k] = B(k0+k, n0+n) ----
        if (btrans) {
            if (fb && inK) {
                #pragma unroll
                for (int i = 0; i < 4; ++i) {
                    int idx = tid + i * 256;
                    int n = idx >> 3, cg = (idx & 7) * 8;
                    unsigned laddr = (unsigned)(unsigned long long)&BsT[buf][n][cg];
                    const bf16_t* ga = &Bm[(long long)(n0 + n) * ldb + k0 + cg];
                    asm volatile("global_load_async_to_lds_b128 %0, %1, off"
                                 :: "v"(laddr), "v"(ga) : "memory");
                }
                issued += 4;
            } else {
                for (int idx = tid; idx < 128 * 64; idx += 256) {
                    int n = idx >> 6, kk = idx & 63;
                    int gn = n0 + n, gk = k0 + kk;
                    bf16_t v = (bf16_t)0.0f;
                    if (gn < N && gk < K) v = Bm[(long long)gn * ldb + gk];
                    BsT[buf][n][kk] = v;
                }
            }
        } else {
            if (fb && inK) {
                #pragma unroll
                for (int i = 0; i < 4; ++i) {
                    int idx = tid + i * 256;
                    int kk = idx >> 4, ng = (idx & 15) * 8;
                    v8bf v = *(const v8bf*)&Bm[(long long)(k0 + kk) * ldb + n0 + ng];
                    #pragma unroll
                    for (int e = 0; e < 8; ++e) BsT[buf][ng + e][kk] = v[e];
                }
            } else {
                for (int idx = tid; idx < 128 * 64; idx += 256) {
                    int n = idx >> 6, kk = idx & 63;
                    int gn = n0 + n, gk = k0 + kk;
                    bf16_t v = (bf16_t)0.0f;
                    if (gn < N && gk < K) v = Bm[(long long)gk * ldb + gn];
                    BsT[buf][n][kk] = v;
                }
            }
        }
        return issued;
    };

    v8f c00 = {}, c01 = {}, c10 = {}, c11 = {};

    (void)stageTile(0, 0);
    int buf = 0;
    for (int k0 = 0; k0 < K; k0 += 64) {
        // stage next tile into the other buffer while current computes
        int pend = 0;
        if (k0 + 64 < K) pend = stageTile(buf ^ 1, k0 + 64);
        // retire the current tile's async loads (in-order completion: leaving
        // `pend` outstanding means everything older is done)
        if (pend >= 6)      asm volatile("s_wait_asynccnt 0x6" ::: "memory");
        else if (pend >= 4) asm volatile("s_wait_asynccnt 0x4" ::: "memory");
        else if (pend >= 2) asm volatile("s_wait_asynccnt 0x2" ::: "memory");
        else                asm volatile("s_wait_asynccnt 0x0" ::: "memory");
        __syncthreads();

        // ---- 2 k-steps of 32 from the staged tile ----
        int ar0 = wm2 + l16, ar1 = ar0 + 16;
        int bc0 = wn2 + l16, bc1 = bc0 + 16;
        #pragma unroll
        for (int kh = 0; kh < 2; ++kh) {
            int kb = kh * 32;
            // A 16x32 frag: lane<16 row=lane K{0..7,16..23}; lane>=16 K{8..15,24..31}
            v8bf a0lo = *(const v8bf*)&As[buf][ar0][kb + half * 8];
            v8bf a0hi = *(const v8bf*)&As[buf][ar0][kb + 16 + half * 8];
            v8bf a1lo = *(const v8bf*)&As[buf][ar1][kb + half * 8];
            v8bf a1hi = *(const v8bf*)&As[buf][ar1][kb + 16 + half * 8];
            v16bf a0 = __builtin_shufflevector(a0lo, a0hi, 0,1,2,3,4,5,6,7,8,9,10,11,12,13,14,15);
            v16bf a1 = __builtin_shufflevector(a1lo, a1hi, 0,1,2,3,4,5,6,7,8,9,10,11,12,13,14,15);
            // B 32x16 frag: col = lane&15, lane<16 K{0..7,16..23}; lane>=16 K{8..15,24..31}
            v8bf b0lo = *(const v8bf*)&BsT[buf][bc0][kb + half * 8];
            v8bf b0hi = *(const v8bf*)&BsT[buf][bc0][kb + 16 + half * 8];
            v8bf b1lo = *(const v8bf*)&BsT[buf][bc1][kb + half * 8];
            v8bf b1hi = *(const v8bf*)&BsT[buf][bc1][kb + 16 + half * 8];
            v16bf b0 = __builtin_shufflevector(b0lo, b0hi, 0,1,2,3,4,5,6,7,8,9,10,11,12,13,14,15);
            v16bf b1 = __builtin_shufflevector(b1lo, b1hi, 0,1,2,3,4,5,6,7,8,9,10,11,12,13,14,15);

            c00 = __builtin_amdgcn_wmma_f32_16x16x32_bf16(false, a0, false, b0, (short)0, c00, false, false);
            c01 = __builtin_amdgcn_wmma_f32_16x16x32_bf16(false, a0, false, b1, (short)0, c01, false, false);
            c10 = __builtin_amdgcn_wmma_f32_16x16x32_bf16(false, a1, false, b0, (short)0, c10, false, false);
            c11 = __builtin_amdgcn_wmma_f32_16x16x32_bf16(false, a1, false, b1, (short)0, c11, false, false);
        }
        __syncthreads();
        buf ^= 1;
    }

    float sc = scale;
    if (zscale) sc *= zscale[hh];

    #pragma unroll
    for (int mi = 0; mi < 2; ++mi) {
        #pragma unroll
        for (int ni = 0; ni < 2; ++ni) {
            v8f acc = mi ? (ni ? c11 : c10) : (ni ? c01 : c00);
            int gn = n0 + wn2 + ni * 16 + l16;
            #pragma unroll
            for (int r = 0; r < 8; ++r) {
                int gm = m0 + wm2 + mi * 16 + half * 8 + r;
                if (gm < M && gn < N) {
                    float v = acc[r] * sc;
                    if (bias)  v += bias[gn];
                    if (bmat)  v += bmat[(long long)hh * bmat_h + (long long)gm * bmat_ld + gn];
                    if (causal && gn > gm) v = NEG_INF;
                    if (resid) v += resid[coff + (long long)gm * ldc + gn];
                    long long o = coff + (long long)gm * ldc + gn;
                    if (C)   C[o]   = v;
                    if (Cbf) Cbf[o] = (bf16_t)v;
                }
            }
        }
    }
}

// ---------------------------------------------------------------------------
__global__ __launch_bounds__(256)
void cvt_kernel(const float* __restrict__ in, bf16_t* __restrict__ out, long long n)
{
    long long i = (long long)blockIdx.x * 256 + threadIdx.x;
    if (i < n) out[i] = (bf16_t)in[i];
}

__global__ __launch_bounds__(256)
void rmsnorm_kernel(const float* __restrict__ x, const float* __restrict__ g,
                    bf16_t* __restrict__ out, int ncols)
{
    __shared__ float red[8];
    long long row = blockIdx.x;
    const float* xr = x + row * ncols;
    int tid = threadIdx.x;
    float ss = 0.f;
    for (int i = tid; i < ncols; i += 256) { float v = xr[i]; ss = fmaf(v, v, ss); }
    for (int o = 16; o > 0; o >>= 1) ss += __shfl_xor(ss, o);
    if ((tid & 31) == 0) red[tid >> 5] = ss;
    __syncthreads();
    float tot = 0.f;
    #pragma unroll
    for (int w = 0; w < 8; ++w) tot += red[w];
    float inv = rsqrtf(tot / (float)ncols + 1e-8f);
    bf16_t* orow = out + row * ncols;
    for (int i = tid; i < ncols; i += 256) orow[i] = (bf16_t)(xr[i] * inv * g[i]);
}

__global__ __launch_bounds__(256)
void softmax_kernel(const float* __restrict__ dots, bf16_t* __restrict__ attn,
                    int S, int SPAD)
{
    __shared__ float red[8];
    long long row = blockIdx.x;
    const float* dr = dots + row * SPAD;
    bf16_t*      ar = attn + row * SPAD;
    int tid = threadIdx.x;
    float mx = NEG_INF;
    for (int j = tid; j < S; j += 256) mx = fmaxf(mx, dr[j]);
    for (int o = 16; o > 0; o >>= 1) mx = fmaxf(mx, __shfl_xor(mx, o));
    if ((tid & 31) == 0) red[tid >> 5] = mx;
    __syncthreads();
    float m2 = NEG_INF;
    #pragma unroll
    for (int w = 0; w < 8; ++w) m2 = fmaxf(m2, red[w]);
    __syncthreads();
    float sum = 0.f;
    for (int j = tid; j < S; j += 256) sum += __expf(dr[j] - m2);
    for (int o = 16; o > 0; o >>= 1) sum += __shfl_xor(sum, o);
    if ((tid & 31) == 0) red[tid >> 5] = sum;
    __syncthreads();
    float tot = 0.f;
    #pragma unroll
    for (int w = 0; w < 8; ++w) tot += red[w];
    float inv = 1.f / tot;
    for (int j = tid; j < SPAD; j += 256)
        ar[j] = (j < S) ? (bf16_t)(__expf(dr[j] - m2) * inv) : (bf16_t)0.0f;
}

__global__ __launch_bounds__(256)
void geglu_kernel(const float* __restrict__ h2, bf16_t* __restrict__ out, long long total)
{
    long long idx = (long long)blockIdx.x * 256 + threadIdx.x;   // rows*4096
    if (idx >= total) return;
    long long r = idx >> 12;
    int j = (int)(idx & 4095);
    float a = h2[r * 8192 + j];
    float g = h2[r * 8192 + 4096 + j];
    float t = tanhf(0.7978845608f * (g + 0.044715f * g * g * g));
    out[idx] = (bf16_t)(a * 0.5f * g * (1.f + t));
}

// T5 relative-position bias, scaled by sqrt(DH)=8
__global__ __launch_bounds__(256)
void relbias_kernel(const float* __restrict__ table, float* __restrict__ out,
                    int S, int causal)
{
    int idx = blockIdx.x * 256 + threadIdx.x;
    if (idx >= S * S) return;
    int i = idx / S, j = idx % S;
    int n = i - j;                     // n = -rel
    int bucket;
    if (causal) {
        if (n < 0) n = 0;
        if (n < 16) bucket = n;
        else {
            int vl = 16 + (int)(logf((float)n / 16.f) / logf(8.f) * 16.f);
            bucket = vl < 31 ? vl : 31;
        }
    } else {
        int ret = (n < 0) ? 16 : 0;
        int na = n < 0 ? -n : n;
        if (na < 8) bucket = ret + na;
        else {
            int vl = 8 + (int)(logf((float)na / 8.f) / logf(16.f) * 8.f);
            bucket = ret + (vl < 15 ? vl : 15);
        }
    }
    for (int hh = 0; hh < 16; ++hh)
        out[(long long)hh * S * S + idx] = table[bucket * 16 + hh] * 8.0f;
}

__global__ __launch_bounds__(256)
void enc_prep_kernel(float* __restrict__ h, const float* __restrict__ mt,
                     const float* __restrict__ lt)
{
    int idx = blockIdx.x * 256 + threadIdx.x;   // 32*1024
    if (idx >= 32 * 1024) return;
    int bb = idx >> 10, c = idx & 1023;
    h[(long long)bb * 198 * 1024 + c]        = mt[c];
    h[(long long)bb * 198 * 1024 + 1024 + c] = lt[c];
}

__global__ __launch_bounds__(256)
void take2_kernel(const float* __restrict__ h, float* __restrict__ out)
{
    int idx = blockIdx.x * 256 + threadIdx.x;   // 64*1024
    if (idx >= 64 * 1024) return;
    int row = idx >> 10, c = idx & 1023;
    int bb = row >> 1, r = row & 1;
    out[idx] = h[(long long)bb * 198 * 1024 + r * 1024 + c];
}

__global__ __launch_bounds__(256)
void reparam_kernel(const float* __restrict__ muall, const float* __restrict__ lvall,
                    const float* __restrict__ eps, float* __restrict__ out_mu,
                    float* __restrict__ out_lv, bf16_t* __restrict__ zbf)
{
    int idx = blockIdx.x * 256 + threadIdx.x;   // 32*256
    if (idx >= 32 * 256) return;
    int bb = idx >> 8, i = idx & 255;
    float mu = muall[(2 * bb) * 256 + i];
    float lv = lvall[(2 * bb + 1) * 256 + i];
    out_mu[idx] = mu;
    out_lv[idx] = lv;
    zbf[idx] = (bf16_t)(mu + eps[idx] * __expf(0.5f * lv));
}

__global__ __launch_bounds__(256)
void vscale_kernel(const float* __restrict__ v, const float* __restrict__ hs,
                   bf16_t* __restrict__ out)
{
    int idx = blockIdx.x * 256 + threadIdx.x;   // 32*1024
    if (idx >= 32 * 1024) return;
    out[idx] = (bf16_t)(v[idx] * hs[(idx & 1023) >> 6]);
}

__global__ __launch_bounds__(256)
void bcast_rows_kernel(const float* __restrict__ src, float* __restrict__ dst,
                       long long rows_per_b, int nb)
{
    long long idx = (long long)blockIdx.x * 256 + threadIdx.x;
    long long total = (long long)nb * rows_per_b;
    if (idx >= total) return;
    dst[idx] = src[idx % rows_per_b];
}

__global__ __launch_bounds__(256)
void cross_add_kernel(float* __restrict__ d, const float* __restrict__ crow)
{
    long long idx = (long long)blockIdx.x * 256 + threadIdx.x; // 32*196*1024
    if (idx >= (long long)32 * 196 * 1024) return;
    int bb = (int)(idx / (196 * 1024));
    int c  = (int)(idx & 1023);
    d[idx] += crow[bb * 1024 + c];
}

// ---------------------------------------------------------------------------
static void launch_gemm(hipStream_t s, const bf16_t* A, const bf16_t* Bm,
                        float* C, bf16_t* Cbf, const float* bias,
                        const float* bmat, int bmat_ld, long long bmat_h,
                        const float* resid, const float* zscale,
                        int M, int N, int K, int lda, int ldb, int ldc,
                        long long sAb, long long sAh, long long sBb, long long sBh,
                        long long sCb, long long sCh, int Z, int zmod,
                        float scale, int btrans, int causal)
{
    dim3 g((N + 127) / 128, (M + 63) / 64, Z);
    wmma_gemm_kernel<<<g, dim3(256), 0, s>>>(A, Bm, C, Cbf, bias, bmat, bmat_ld, bmat_h,
                                             resid, zscale, M, N, K, lda, ldb, ldc,
                                             sAb, sAh, sBb, sBh, sCb, sCh, zmod,
                                             scale, btrans, causal);
}

static void launch_gemm1(hipStream_t s, const bf16_t* A, const bf16_t* Bm,
                         float* C, bf16_t* Cbf, const float* bias,
                         const float* resid, int M, int N, int K)
{
    launch_gemm(s, A, Bm, C, Cbf, bias, nullptr, 0, 0, resid, nullptr,
                M, N, K, K, N, N, 0, 0, 0, 0, 0, 0, 1, 1, 1.0f, 0, 0);
}

// ---------------------------------------------------------------------------
extern "C" void kernel_launch(void* const* d_in, const int* in_sizes, int n_in,
                              void* d_out, int out_size, void* d_ws, size_t ws_size,
                              hipStream_t stream)
{
    (void)in_sizes; (void)n_in; (void)out_size; (void)ws_size;
    constexpr int Bn = 32, SEQ = 196, POSE = 263, LAT = 256, Hd = 1024;
    constexpr int HEADS = 16, LAYERS = 4, INNER = 4096;
    constexpr int SE = SEQ + 2;      // 198
    constexpr int SPAD = 224;        // 7*32 K-padding for attention GEMMs

    auto Pf = [&](int i) { return (const float*)d_in[i]; };
    // --- input index map (recursive dict insertion order) ---
    const int I_x = 0, I_eps = 1, I_enc_in_w = 2, I_enc_in_b = 3, I_mu_tok = 4, I_lv_tok = 5;
    auto EL = [](int l, int o) { return 6 + 12 * l + o; };     // norm_g,wq,wk,wv,hs,ow,ob,fnorm,w1w,w1b,w2w,w2b
    const int I_enc_relpos = 54, I_enc_final_g = 55, I_enc_out_w = 56, I_enc_out_b = 57;
    const int I_mu_w = 58, I_mu_b = 59, I_lv_w = 60, I_lv_b = 61, I_l2h_w = 62, I_l2h_b = 63;
    const int I_query = 64, I_dec_in_w = 65, I_dec_in_b = 66;
    auto DL = [](int l, int o) { return 67 + 19 * l + o; };    // attn(7), cross(7), ff(5)
    const int I_dec_relpos = 143, I_dec_final_g = 144, I_dec_out_w = 145, I_dec_out_b = 146;
    const int I_out_w = 147, I_out_b = 148;

    // --- workspace bump allocator ---
    char* base = (char*)d_ws;
    size_t cur = 0;
    auto allocB = [&](size_t bytes) -> void* {
        cur = (cur + 255) & ~(size_t)255;
        void* p = base + cur; cur += bytes; return p;
    };
    auto aF = [&](long long n) { return (float*)allocB((size_t)n * 4); };
    auto aH = [&](long long n) { return (bf16_t*)allocB((size_t)n * 2); };

    auto cvt = [&](int idx, bf16_t* dst, long long n) {
        cvt_kernel<<<dim3((unsigned)((n + 255) / 256)), dim3(256), 0, stream>>>(Pf(idx), dst, n);
    };

    // bf16 weights / constant inputs
    bf16_t* xbf = aH((long long)Bn * SEQ * POSE);
    bf16_t* w_enc_in = aH((long long)POSE * Hd);
    bf16_t *e_wq[LAYERS], *e_wk[LAYERS], *e_wv[LAYERS], *e_wo[LAYERS], *e_w1[LAYERS], *e_w2[LAYERS];
    for (int l = 0; l < LAYERS; ++l) {
        e_wq[l] = aH((long long)Hd * Hd); e_wk[l] = aH((long long)Hd * Hd);
        e_wv[l] = aH((long long)Hd * Hd); e_wo[l] = aH((long long)Hd * Hd);
        e_w1[l] = aH((long long)Hd * 2 * INNER); e_w2[l] = aH((long long)INNER * Hd);
    }
    bf16_t* w_enc_out = aH((long long)Hd * Hd);
    bf16_t* w_mu  = aH((long long)Hd * LAT);
    bf16_t* w_lv  = aH((long long)Hd * LAT);
    bf16_t* w_l2h = aH((long long)LAT * Hd);
    bf16_t* qemb  = aH((long long)SEQ * Hd);
    bf16_t* w_dec_in = aH((long long)Hd * Hd);
    bf16_t *d_wq[LAYERS], *d_wk[LAYERS], *d_wv[LAYERS], *d_wo[LAYERS],
           *d_cwv[LAYERS], *d_cwo[LAYERS], *d_w1[LAYERS], *d_w2[LAYERS];
    for (int l = 0; l < LAYERS; ++l) {
        d_wq[l] = aH((long long)Hd * Hd); d_wk[l] = aH((long long)Hd * Hd);
        d_wv[l] = aH((long long)Hd * Hd); d_wo[l] = aH((long long)Hd * Hd);
        d_cwv[l] = aH((long long)Hd * Hd); d_cwo[l] = aH((long long)Hd * Hd);
        d_w1[l] = aH((long long)Hd * 2 * INNER); d_w2[l] = aH((long long)INNER * Hd);
    }
    bf16_t* w_dec_out = aH((long long)Hd * Hd);
    bf16_t* w_out = aH((long long)Hd * POSE);

    // activations (sized for max rows = 32*198)
    const long long RE = (long long)Bn * SE;     // 6336
    const long long RD = (long long)Bn * SEQ;    // 6272
    float*  hstm = aF(RE * Hd);
    float*  dstm = aF(RD * Hd);
    bf16_t* hnbf = aH(RE * Hd);
    bf16_t* qb = aH(RE * Hd);  bf16_t* kb = aH(RE * Hd);  bf16_t* vb = aH(RE * Hd);
    float*  dots = aF((long long)Bn * HEADS * SE * SPAD);
    bf16_t* attb = aH((long long)Bn * HEADS * SE * SPAD);
    bf16_t* ob = aH(RE * Hd);
    float*  h2 = aF(RE * 2 * INNER);
    bf16_t* gb = aH(RE * INNER);
    float*  biasE = aF((long long)HEADS * SE * SE);
    float*  biasD = aF((long long)HEADS * SEQ * SEQ);
    float*  rows01 = aF(64 * Hd);
    bf16_t* he2 = aH(64 * Hd);
    bf16_t* hob = aH(64 * Hd);
    float*  muall = aF(64 * LAT);
    float*  lvall = aF(64 * LAT);
    bf16_t* zb = aH((long long)Bn * LAT);
    bf16_t* ctxb = aH((long long)Bn * Hd);
    float*  vctx = aF((long long)Bn * Hd);
    bf16_t* vsb = aH((long long)Bn * Hd);
    float*  crow = aF((long long)Bn * Hd);
    float*  tmpd = aF((long long)SEQ * Hd);

    float* out_recon = (float*)d_out;
    float* out_mu = out_recon + (long long)Bn * SEQ * POSE;
    float* out_lv = out_mu + (long long)Bn * LAT;

    // ---- weight conversion ----
    cvt(I_x, xbf, (long long)Bn * SEQ * POSE);
    cvt(I_enc_in_w, w_enc_in, (long long)POSE * Hd);
    for (int l = 0; l < LAYERS; ++l) {
        cvt(EL(l, 1), e_wq[l], (long long)Hd * Hd);
        cvt(EL(l, 2), e_wk[l], (long long)Hd * Hd);
        cvt(EL(l, 3), e_wv[l], (long long)Hd * Hd);
        cvt(EL(l, 5), e_wo[l], (long long)Hd * Hd);
        cvt(EL(l, 8), e_w1[l], (long long)Hd * 2 * INNER);
        cvt(EL(l, 10), e_w2[l], (long long)INNER * Hd);
    }
    cvt(I_enc_out_w, w_enc_out, (long long)Hd * Hd);
    cvt(I_mu_w, w_mu, (long long)Hd * LAT);
    cvt(I_lv_w, w_lv, (long long)Hd * LAT);
    cvt(I_l2h_w, w_l2h, (long long)LAT * Hd);
    cvt(I_query, qemb, (long long)SEQ * Hd);
    cvt(I_dec_in_w, w_dec_in, (long long)Hd * Hd);
    for (int l = 0; l < LAYERS; ++l) {
        cvt(DL(l, 1), d_wq[l], (long long)Hd * Hd);
        cvt(DL(l, 2), d_wk[l], (long long)Hd * Hd);
        cvt(DL(l, 3), d_wv[l], (long long)Hd * Hd);
        cvt(DL(l, 5), d_wo[l], (long long)Hd * Hd);
        cvt(DL(l, 10), d_cwv[l], (long long)Hd * Hd);
        cvt(DL(l, 12), d_cwo[l], (long long)Hd * Hd);
        cvt(DL(l, 15), d_w1[l], (long long)Hd * 2 * INNER);
        cvt(DL(l, 17), d_w2[l], (long long)INNER * Hd);
    }
    cvt(I_dec_out_w, w_dec_out, (long long)Hd * Hd);
    cvt(I_out_w, w_out, (long long)Hd * POSE);

    // ---- relative position bias buffers ----
    relbias_kernel<<<dim3((SE * SE + 255) / 256), dim3(256), 0, stream>>>(Pf(I_enc_relpos), biasE, SE, 0);
    relbias_kernel<<<dim3((SEQ * SEQ + 255) / 256), dim3(256), 0, stream>>>(Pf(I_dec_relpos), biasD, SEQ, 1);

    // =========================== ENCODER ===========================
    // h[:,2:,:] = x @ enc_in.w + b  (batched over b, row offset 2)
    launch_gemm(stream, xbf, w_enc_in, hstm + 2 * Hd, nullptr, Pf(I_enc_in_b),
                nullptr, 0, 0, nullptr, nullptr,
                SEQ, Hd, POSE, POSE, Hd, Hd,
                (long long)SEQ * POSE, 0, 0, 0, (long long)SE * Hd, 0, Bn, 1, 1.0f, 0, 0);
    enc_prep_kernel<<<dim3(128), dim3(256), 0, stream>>>(hstm, Pf(I_mu_tok), Pf(I_lv_tok));

    for (int l = 0; l < LAYERS; ++l) {
        rmsnorm_kernel<<<dim3((unsigned)RE), dim3(256), 0, stream>>>(hstm, Pf(EL(l, 0)), hnbf, Hd);
        launch_gemm1(stream, hnbf, e_wq[l], nullptr, qb, nullptr, nullptr, (int)RE, Hd, Hd);
        launch_gemm1(stream, hnbf, e_wk[l], nullptr, kb, nullptr, nullptr, (int)RE, Hd, Hd);
        launch_gemm1(stream, hnbf, e_wv[l], nullptr, vb, nullptr, nullptr, (int)RE, Hd, Hd);
        // dots = 0.125 * Q K^T + bias   [Z=512, b=z/16, h=z%16]
        launch_gemm(stream, qb, kb, dots, nullptr, nullptr,
                    biasE, SE, (long long)SE * SE, nullptr, nullptr,
                    SE, SE, 64, Hd, Hd, SPAD,
                    (long long)SE * Hd, 64, (long long)SE * Hd, 64,
                    (long long)HEADS * SE * SPAD, (long long)SE * SPAD,
                    Bn * HEADS, HEADS, 0.125f, 1, 0);
        softmax_kernel<<<dim3((unsigned)(Bn * HEADS * SE)), dim3(256), 0, stream>>>(dots, attb, SE, SPAD);
        // O = P V, scaled by head_scale[h]
        launch_gemm(stream, attb, vb, nullptr, ob, nullptr,
                    nullptr, 0, 0, nullptr, Pf(EL(l, 4)),
                    SE, 64, SE, SPAD, Hd, Hd,
                    (long long)HEADS * SE * SPAD, (long long)SE * SPAD,
                    (long long)SE * Hd, 64, (long long)SE * Hd, 64,
                    Bn * HEADS, HEADS, 1.0f, 0, 0);
        launch_gemm1(stream, ob, e_wo[l], hstm, nullptr, Pf(EL(l, 6)), hstm, (int)RE, Hd, Hd);
        // GEGLU FF
        rmsnorm_kernel<<<dim3((unsigned)RE), dim3(256), 0, stream>>>(hstm, Pf(EL(l, 7)), hnbf, Hd);
        launch_gemm1(stream, hnbf, e_w1[l], h2, nullptr, Pf(EL(l, 9)), nullptr, (int)RE, 2 * INNER, Hd);
        geglu_kernel<<<dim3((unsigned)((RE * INNER + 255) / 256)), dim3(256), 0, stream>>>(h2, gb, RE * INNER);
        launch_gemm1(stream, gb, e_w2[l], hstm, nullptr, Pf(EL(l, 11)), hstm, (int)RE, Hd, INNER);
    }

    // final norm + enc_out on tokens 0,1 only; mu/logvar; reparam; lat2hid
    take2_kernel<<<dim3(256), dim3(256), 0, stream>>>(hstm, rows01);
    rmsnorm_kernel<<<dim3(64), dim3(256), 0, stream>>>(rows01, Pf(I_enc_final_g), he2, Hd);
    launch_gemm1(stream, he2, w_enc_out, nullptr, hob, Pf(I_enc_out_b), nullptr, 64, Hd, Hd);
    launch_gemm1(stream, hob, w_mu, muall, nullptr, Pf(I_mu_b), nullptr, 64, LAT, Hd);
    launch_gemm1(stream, hob, w_lv, lvall, nullptr, Pf(I_lv_b), nullptr, 64, LAT, Hd);
    reparam_kernel<<<dim3(32), dim3(256), 0, stream>>>(muall, lvall, Pf(I_eps), out_mu, out_lv, zb);
    launch_gemm1(stream, zb, w_l2h, nullptr, ctxb, Pf(I_l2h_b), nullptr, Bn, Hd, LAT);

    // =========================== DECODER ===========================
    launch_gemm1(stream, qemb, w_dec_in, tmpd, nullptr, Pf(I_dec_in_b), nullptr, SEQ, Hd, Hd);
    bcast_rows_kernel<<<dim3((unsigned)((RD * Hd + 255) / 256)), dim3(256), 0, stream>>>(
        tmpd, dstm, (long long)SEQ * Hd, Bn);

    for (int l = 0; l < LAYERS; ++l) {
        rmsnorm_kernel<<<dim3((unsigned)RD), dim3(256), 0, stream>>>(dstm, Pf(DL(l, 0)), hnbf, Hd);
        launch_gemm1(stream, hnbf, d_wq[l], nullptr, qb, nullptr, nullptr, (int)RD, Hd, Hd);
        launch_gemm1(stream, hnbf, d_wk[l], nullptr, kb, nullptr, nullptr, (int)RD, Hd, Hd);
        launch_gemm1(stream, hnbf, d_wv[l], nullptr, vb, nullptr, nullptr, (int)RD, Hd, Hd);
        launch_gemm(stream, qb, kb, dots, nullptr, nullptr,
                    biasD, SEQ, (long long)SEQ * SEQ, nullptr, nullptr,
                    SEQ, SEQ, 64, Hd, Hd, SPAD,
                    (long long)SEQ * Hd, 64, (long long)SEQ * Hd, 64,
                    (long long)HEADS * SEQ * SPAD, (long long)SEQ * SPAD,
                    Bn * HEADS, HEADS, 0.125f, 1, /*causal=*/1);
        softmax_kernel<<<dim3((unsigned)(Bn * HEADS * SEQ)), dim3(256), 0, stream>>>(dots, attb, SEQ, SPAD);
        launch_gemm(stream, attb, vb, nullptr, ob, nullptr,
                    nullptr, 0, 0, nullptr, Pf(DL(l, 4)),
                    SEQ, 64, SEQ, SPAD, Hd, Hd,
                    (long long)HEADS * SEQ * SPAD, (long long)SEQ * SPAD,
                    (long long)SEQ * Hd, 64, (long long)SEQ * Hd, 64,
                    Bn * HEADS, HEADS, 1.0f, 0, 0);
        launch_gemm1(stream, ob, d_wo[l], dstm, nullptr, Pf(DL(l, 6)), dstm, (int)RD, Hd, Hd);
        // cross-attn to single latent token: softmax==1 -> o = head_scale * (ctx@Wv)
        launch_gemm1(stream, ctxb, d_cwv[l], vctx, nullptr, nullptr, nullptr, Bn, Hd, Hd);
        vscale_kernel<<<dim3(128), dim3(256), 0, stream>>>(vctx, Pf(DL(l, 11)), vsb);
        launch_gemm1(stream, vsb, d_cwo[l], crow, nullptr, Pf(DL(l, 13)), nullptr, Bn, Hd, Hd);
        cross_add_kernel<<<dim3((unsigned)((RD * Hd + 255) / 256)), dim3(256), 0, stream>>>(dstm, crow);
        // GEGLU FF
        rmsnorm_kernel<<<dim3((unsigned)RD), dim3(256), 0, stream>>>(dstm, Pf(DL(l, 14)), hnbf, Hd);
        launch_gemm1(stream, hnbf, d_w1[l], h2, nullptr, Pf(DL(l, 16)), nullptr, (int)RD, 2 * INNER, Hd);
        geglu_kernel<<<dim3((unsigned)((RD * INNER + 255) / 256)), dim3(256), 0, stream>>>(h2, gb, RD * INNER);
        launch_gemm1(stream, gb, d_w2[l], dstm, nullptr, Pf(DL(l, 18)), dstm, (int)RD, Hd, INNER);
    }

    rmsnorm_kernel<<<dim3((unsigned)RD), dim3(256), 0, stream>>>(dstm, Pf(I_dec_final_g), hnbf, Hd);
    launch_gemm1(stream, hnbf, w_dec_out, nullptr, ob, Pf(I_dec_out_b), nullptr, (int)RD, Hd, Hd);
    launch_gemm1(stream, ob, w_out, out_recon, nullptr, Pf(I_out_b), nullptr, (int)RD, POSE, Hd);
}
